// ConvRecLayer_12266426597625
// MI455X (gfx1250) — matrix operand order repaired and checked by reading, not compiled, not run on hardware
//
#include <hip/hip_runtime.h>
#include <hip/hip_bf16.h>

// Problem constants (from reference)
#define TDIM   2048
#define BDIM   16
#define CDIM   1024
#define NHEAD  16
#define KSZ    15
#define HK     240        // NHEAD*KSZ
#define FFND   4096
#define TBROWS 32768      // TDIM*BDIM
#define EPSF   1e-5f

typedef __attribute__((ext_vector_type(16))) __bf16 bf16x16;
typedef __attribute__((ext_vector_type(8)))  __bf16 bf16x8;
typedef __attribute__((ext_vector_type(4)))  __bf16 bf16x4;
typedef __attribute__((ext_vector_type(8)))  float  v8f;
typedef __attribute__((ext_vector_type(4)))  int    i32x4;
typedef __attribute__((address_space(1)))    i32x4  gas_i32x4;  // global
typedef __attribute__((address_space(3)))    i32x4  las_i32x4;  // LDS

// ---- CDNA5 async global->LDS copy (ASYNCcnt path), with guarded fallback ----
#if defined(__has_builtin)
# if __has_builtin(__builtin_amdgcn_global_load_async_to_lds_b128)
#  define HAVE_ASYNC_LDS 1
# endif
#endif
#ifndef HAVE_ASYNC_LDS
# define HAVE_ASYNC_LDS 0
#endif

__device__ __forceinline__ void async_cp16(const __bf16* g, __bf16* l) {
#if HAVE_ASYNC_LDS
    __builtin_amdgcn_global_load_async_to_lds_b128(
        (gas_i32x4*)g, (las_i32x4*)l, 0, 0);
#else
    *(uint4*)l = *(const uint4*)g;
#endif
}

template <int N>
__device__ __forceinline__ void wait_async() {
#if HAVE_ASYNC_LDS
# if defined(__has_builtin) && __has_builtin(__builtin_amdgcn_s_wait_asynccnt)
    __builtin_amdgcn_s_wait_asynccnt(N);
# else
    if constexpr (N == 0) asm volatile("s_wait_asynccnt 0x0" ::: "memory");
    else                  asm volatile("s_wait_asynccnt 0x4" ::: "memory");
# endif
#endif
}

__device__ __forceinline__ float4 ld4(const float* p) { return *(const float4*)p; }

__device__ __forceinline__ bf16x8 cvt8(float4 a, float4 b) {
    bf16x8 r;
    r[0] = (__bf16)a.x; r[1] = (__bf16)a.y; r[2] = (__bf16)a.z; r[3] = (__bf16)a.w;
    r[4] = (__bf16)b.x; r[5] = (__bf16)b.y; r[6] = (__bf16)b.z; r[7] = (__bf16)b.w;
    return r;
}
__device__ __forceinline__ bf16x16 join16(bf16x8 lo, bf16x8 hi) {
    bf16x16 r;
#pragma unroll
    for (int i = 0; i < 8; ++i) { r[i] = lo[i]; r[i + 8] = hi[i]; }
    return r;
}

// ---------------------------------------------------------------------------
// Kernel 0: one-time f32 -> bf16 conversion (conv_w, fc1_w, fc2_w)
// ---------------------------------------------------------------------------
__global__ __launch_bounds__(256)
void cvt_bf16_kernel(const float* __restrict__ in, __bf16* __restrict__ out, int n4) {
    const int i = blockIdx.x * 256 + threadIdx.x;
    if (i < n4) {
        const float4 v = ld4(in + 4 * (size_t)i);
        bf16x4 o;
        o[0] = (__bf16)v.x; o[1] = (__bf16)v.y; o[2] = (__bf16)v.z; o[3] = (__bf16)v.w;
        *(bf16x4*)(out + 4 * (size_t)i) = o;
    }
}

// ---------------------------------------------------------------------------
// Kernel 1: conv weight projection GEMM (TBROWS x 1024) x (1024 x 240)
//           + bias + softmax over the 15 taps of each head.
// One wave per 16-row tile; 15 column tiles held in registers (120 VGPRs).
// ---------------------------------------------------------------------------
__global__ __launch_bounds__(32)
void proj_softmax_kernel(const float*  __restrict__ x,
                         const __bf16* __restrict__ cwb,  // [240][1024] bf16
                         const float*  __restrict__ cb,   // [240]
                         float*        __restrict__ wout) // [TBROWS][240]
{
    __shared__ float sw[16][HK];

    const int lane = threadIdx.x;
    const int half = lane >> 4;
    const int ml   = lane & 15;
    const int row0 = blockIdx.x * 16;

    v8f acc[15];
#pragma unroll
    for (int nt = 0; nt < 15; ++nt) acc[nt] = (v8f)(0.f);

    for (int kb = 0; kb < CDIM; kb += 32) {
        // A fragment: M = ml, k in {8*half..+7} U {16+8*half..+7}
        const float* ap = x + (size_t)(row0 + ml) * CDIM + kb + 8 * half;
        bf16x16 afr = join16(cvt8(ld4(ap),      ld4(ap + 4)),
                             cvt8(ld4(ap + 16), ld4(ap + 20)));
#pragma unroll
        for (int nt = 0; nt < 15; ++nt) {
            // B fragment: n = nt*16+ml, k in {8*half..+7} U {16+8*half..+7}
            const __bf16* wp = cwb + (size_t)(nt * 16 + ml) * CDIM + kb + 8 * half;
            bf16x16 bfr = join16(*(const bf16x8*)wp, *(const bf16x8*)(wp + 16));
            acc[nt] = __builtin_amdgcn_wmma_f32_16x16x32_bf16(
                false, afr, false, bfr, (short)0, acc[nt], false, false);
        }
    }

    // Spill to LDS with bias (C/D layout: m = v + 8*half, n = ml)
#pragma unroll
    for (int nt = 0; nt < 15; ++nt) {
        const int ncol = nt * 16 + ml;
        const float bs = cb[ncol];
#pragma unroll
        for (int v = 0; v < 8; ++v)
            sw[v + 8 * half][ncol] = acc[nt][v] + bs;
    }
    __syncthreads();

    // Softmax over 15 taps: 16 rows * 16 heads = 256 items, 8 per lane
#pragma unroll
    for (int i = 0; i < 8; ++i) {
        const int id = lane + 32 * i;
        const int r  = id >> 4;
        const int hh = id & 15;
        float vals[KSZ];
        float mx = -1e30f;
#pragma unroll
        for (int k = 0; k < KSZ; ++k) {
            vals[k] = sw[r][hh * KSZ + k];
            mx = fmaxf(mx, vals[k]);
        }
        float s = 0.f;
#pragma unroll
        for (int k = 0; k < KSZ; ++k) { vals[k] = __expf(vals[k] - mx); s += vals[k]; }
        const float inv = 1.f / s;
        float* op = wout + (size_t)(row0 + r) * HK + hh * KSZ;
#pragma unroll
        for (int k = 0; k < KSZ; ++k) op[k] = vals[k] * inv;
    }
}

// ---------------------------------------------------------------------------
// Kernel 2: causal dynamic depthwise conv (K=15) + LayerNorm
// -> y (f32, residual) and yb (bf16, GEMM operand)
// ---------------------------------------------------------------------------
__global__ __launch_bounds__(256)
void conv_ln_kernel(const float* __restrict__ x,
                    const float* __restrict__ wsm,  // [TBROWS][240] softmaxed
                    const float* __restrict__ g,
                    const float* __restrict__ bb,
                    float* __restrict__ y,
                    __bf16* __restrict__ yb)
{
    __shared__ float wl[HK];
    __shared__ float rs[256];
    __shared__ float rq[256];

    const int row = blockIdx.x;           // row = t*BDIM + b
    const int t   = row >> 4;
    const int b   = row & 15;
    const int tid = threadIdx.x;
    const int c0  = tid * 4;
    const int hh  = c0 >> 6;              // head index (headdim=64)

    if (tid < HK) wl[tid] = wsm[(size_t)row * HK + tid];
    __syncthreads();

    float a0 = 0.f, a1 = 0.f, a2 = 0.f, a3 = 0.f;
#pragma unroll
    for (int k = 0; k < KSZ; ++k) {
        const int tin = t + k - (KSZ - 1);
        if (tin >= 0) {
            const float wk = wl[hh * KSZ + k];
            const float4 xv = ld4(x + ((size_t)tin * BDIM + b) * CDIM + c0);
            a0 += wk * xv.x; a1 += wk * xv.y; a2 += wk * xv.z; a3 += wk * xv.w;
        }
    }

    rs[tid] = a0 + a1 + a2 + a3;
    rq[tid] = a0 * a0 + a1 * a1 + a2 * a2 + a3 * a3;
    __syncthreads();
#pragma unroll
    for (int st = 128; st > 0; st >>= 1) {
        if (tid < st) { rs[tid] += rs[tid + st]; rq[tid] += rq[tid + st]; }
        __syncthreads();
    }
    const float mu   = rs[0] * (1.f / CDIM);
    const float var  = rq[0] * (1.f / CDIM) - mu * mu;
    const float rstd = rsqrtf(var + EPSF);

    const float4 gv = ld4(g + c0);
    const float4 bv = ld4(bb + c0);
    float4 o;
    o.x = (a0 - mu) * rstd * gv.x + bv.x;
    o.y = (a1 - mu) * rstd * gv.y + bv.y;
    o.z = (a2 - mu) * rstd * gv.z + bv.z;
    o.w = (a3 - mu) * rstd * gv.w + bv.w;
    *(float4*)(y + (size_t)row * CDIM + c0) = o;

    bf16x4 ob;
    ob[0] = (__bf16)o.x; ob[1] = (__bf16)o.y; ob[2] = (__bf16)o.z; ob[3] = (__bf16)o.w;
    *(bf16x4*)(yb + (size_t)row * CDIM + c0) = ob;
}

// ---------------------------------------------------------------------------
// Kernels 3/4: WMMA GEMM with async global->LDS staging, 128x128 block tile,
// 8 waves, each wave 32x64 (2x4 wmma), K-step 32, 3-deep LDS ring.
// Both operands bf16 row-major: A [M][KD], Bw [NTOT][KD].
// ---------------------------------------------------------------------------
template <int KD, bool RELU_BF16_OUT>
__global__ __launch_bounds__(256)
void ffn_gemm_kernel(const __bf16* __restrict__ A,
                     const __bf16* __restrict__ Bw,
                     const float*  __restrict__ bias,
                     const float*  __restrict__ resid,  // null unless f32-out
                     void*         __restrict__ Outp,   // bf16 or f32
                     int NTOT)
{
    constexpr int LDT  = 40;  // padded LDS row stride (bf16) -> conflict-free
    constexpr int NBUF = 3;   // ring depth: overwrite target last read in previous iter
    __shared__ __attribute__((aligned(16))) __bf16 As[NBUF][128 * LDT];
    __shared__ __attribute__((aligned(16))) __bf16 Bs[NBUF][128 * LDT];

    const int tid  = threadIdx.x;
    const int lane = tid & 31;
    const int w    = tid >> 5;
    const int half = lane >> 4;
    const int ml   = lane & 15;
    const int wm   = (w & 3) * 32;   // wave row offset in block tile
    const int wn   = (w >> 2) * 64;  // wave col offset in block tile
    const int m0   = blockIdx.y * 128;
    const int n0   = blockIdx.x * 128;

    // loader mapping: 128 rows x 32 cols per tile, 16 bf16 per thread (2 x b128)
    const int tr  = tid >> 1;
    const int tcs = (tid & 1) * 16;

    v8f acc[2][4];
#pragma unroll
    for (int i = 0; i < 2; ++i)
#pragma unroll
        for (int j = 0; j < 4; ++j) acc[i][j] = (v8f)(0.f);

    // issue one k-tile (4 async b128 per thread -> ASYNCcnt += 4)
    auto issueTile = [&](int kt, int buf) {
        const int kb = kt * 32;
        const __bf16* ga = A  + (size_t)(m0 + tr) * KD + kb + tcs;
        const __bf16* gb = Bw + (size_t)(n0 + tr) * KD + kb + tcs;
        __bf16* la = &As[buf][tr * LDT + tcs];
        __bf16* lb = &Bs[buf][tr * LDT + tcs];
        async_cp16(ga,     la);
        async_cp16(ga + 8, la + 8);
        async_cp16(gb,     lb);
        async_cp16(gb + 8, lb + 8);
    };

    const int nk = KD / 32;
    issueTile(0, 0);
    issueTile(1, 1);

    for (int it = 0; it < nk; ++it) {
        const int buf = it % NBUF;
        // own tile 'it' complete (leave tile it+1 in flight if it exists)
        if (it + 1 < nk) wait_async<4>(); else wait_async<0>();
        __syncthreads();   // all waves: tile 'it' visible, compute(it-1) done
        if (it + 2 < nk) issueTile(it + 2, (it + 2) % NBUF);

        bf16x16 af[2], bfr[4];
#pragma unroll
        for (int i = 0; i < 2; ++i) {
            const __bf16* base = &As[buf][(wm + i * 16 + ml) * LDT];
            af[i] = join16(*(const bf16x8*)(base + 8 * half),
                           *(const bf16x8*)(base + 16 + 8 * half));
        }
#pragma unroll
        for (int j = 0; j < 4; ++j) {
            const __bf16* base = &Bs[buf][(wn + j * 16 + ml) * LDT + 8 * half];
            bfr[j] = join16(*(const bf16x8*)(base),
                            *(const bf16x8*)(base + 16));
        }
#pragma unroll
        for (int i = 0; i < 2; ++i)
#pragma unroll
            for (int j = 0; j < 4; ++j)
                acc[i][j] = __builtin_amdgcn_wmma_f32_16x16x32_bf16(
                    false, af[i], false, bfr[j], (short)0, acc[i][j], false, false);
    }

    // epilogue (C/D layout: m = v + 8*half, n = ml)
#pragma unroll
    for (int i = 0; i < 2; ++i) {
#pragma unroll
        for (int j = 0; j < 4; ++j) {
            const int ncol = n0 + wn + j * 16 + ml;
            const float bs = bias[ncol];
#pragma unroll
            for (int v = 0; v < 8; ++v) {
                const int m = m0 + wm + i * 16 + v + 8 * half;
                const size_t idx = (size_t)m * NTOT + ncol;
                float val = acc[i][j][v] + bs;
                if constexpr (RELU_BF16_OUT) {
                    ((__bf16*)Outp)[idx] = (__bf16)fmaxf(val, 0.f);
                } else {
                    ((float*)Outp)[idx] = val + resid[idx];
                }
            }
        }
    }
}

// ---------------------------------------------------------------------------
// Host launch
// ---------------------------------------------------------------------------
extern "C" void kernel_launch(void* const* d_in, const int* in_sizes, int n_in,
                              void* d_out, int out_size, void* d_ws, size_t ws_size,
                              hipStream_t stream) {
    const float* x      = (const float*)d_in[0];
    const float* conv_w = (const float*)d_in[1];
    const float* conv_b = (const float*)d_in[2];
    const float* ln_g   = (const float*)d_in[3];
    const float* ln_b   = (const float*)d_in[4];
    const float* fc1_w  = (const float*)d_in[5];
    const float* fc1_b  = (const float*)d_in[6];
    const float* fc2_w  = (const float*)d_in[7];
    const float* fc2_b  = (const float*)d_in[8];
    float* out = (float*)d_out;

    char* wsb = (char*)d_ws;
    const size_t W_OFF   = 0;                                       // 31.5 MB
    const size_t Y_OFF   = W_OFF   + (size_t)TBROWS * HK * 4;       // +128 MB
    const size_t YB_OFF  = Y_OFF   + (size_t)TBROWS * CDIM * 4;     // +64  MB
    const size_t H_OFF   = YB_OFF  + (size_t)TBROWS * CDIM * 2;     // +64  MB
    const size_t CWB_OFF = H_OFF   + (size_t)8192 * FFND * 2;       // +0.5 MB
    const size_t F1B_OFF = CWB_OFF + (size_t)HK * CDIM * 2;         // +8   MB
    const size_t F2B_OFF = F1B_OFF + (size_t)FFND * CDIM * 2;       // +8   MB
    float*  w_soft = (float*)(wsb + W_OFF);
    float*  y      = (float*)(wsb + Y_OFF);
    __bf16* yb     = (__bf16*)(wsb + YB_OFF);
    __bf16* hbuf   = (__bf16*)(wsb + H_OFF);
    __bf16* cwb    = (__bf16*)(wsb + CWB_OFF);
    __bf16* f1b    = (__bf16*)(wsb + F1B_OFF);
    __bf16* f2b    = (__bf16*)(wsb + F2B_OFF);

    // 0) one-time weight conversions to bf16
    cvt_bf16_kernel<<<(HK * CDIM / 4 + 255) / 256, 256, 0, stream>>>(
        conv_w, cwb, HK * CDIM / 4);
    cvt_bf16_kernel<<<(FFND * CDIM / 4 + 255) / 256, 256, 0, stream>>>(
        fc1_w, f1b, FFND * CDIM / 4);
    cvt_bf16_kernel<<<(FFND * CDIM / 4 + 255) / 256, 256, 0, stream>>>(
        fc2_w, f2b, FFND * CDIM / 4);

    // 1) projection GEMM + softmax
    proj_softmax_kernel<<<TBROWS / 16, 32, 0, stream>>>(x, cwb, conv_b, w_soft);

    // 2) dynamic conv + LayerNorm -> y (f32) and yb (bf16)
    conv_ln_kernel<<<TBROWS, 256, 0, stream>>>(x, w_soft, ln_g, ln_b, y, yb);

    // 3/4) FFN GEMMs, chunked over rows so h stays at 64 MB bf16
    constexpr int CHUNK_M = 8192;
    for (int ch = 0; ch < TBROWS / CHUNK_M; ++ch) {
        const size_t r0 = (size_t)ch * CHUNK_M;

        dim3 g1(FFND / 128, CHUNK_M / 128);
        ffn_gemm_kernel<CDIM, true><<<g1, 256, 0, stream>>>(
            yb + r0 * CDIM, f1b, fc1_b, nullptr, (void*)hbuf, FFND);

        dim3 g2(CDIM / 128, CHUNK_M / 128);
        ffn_gemm_kernel<FFND, false><<<g2, 256, 0, stream>>>(
            hbuf, f2b, fc2_b, y + r0 * CDIM, (void*)(out + r0 * CDIM), CDIM);
    }
}
